// TemporalFeatureCalibration_89816356094612
// MI455X (gfx1250) — compile-verified, matrix operand order checked by reading, not run on hardware
//
#include <hip/hip_runtime.h>
#include <stdint.h>

// ---------------------------------------------------------------------------
// MI455X (gfx1250) implementation of TemporalFeatureCalibration.
// conv2 (256->256x3x3) and the DCN einsum (~9.7 GMAC each) run as
// M=256 x N=16384 x K=2304 GEMMs on v_wmma_f32_16x16x32_bf16, with
// async global->LDS staging (ASYNCcnt) and b128 LDS fragment loads.
// ---------------------------------------------------------------------------

typedef __attribute__((ext_vector_type(16))) __bf16 bf16x16;
typedef __attribute__((ext_vector_type(8)))  float  f32x8;

union Frag { bf16x16 v; uint32_t u[8]; };

static __device__ __forceinline__ uint16_t f2bf_bits(float f) {
  union { float f; uint32_t u; } x; x.f = f;
  uint32_t r = x.u + 0x7FFFu + ((x.u >> 16) & 1u);   // round-to-nearest-even
  return (uint16_t)(r >> 16);
}

// --------------------- gfx1250 async global->LDS copies --------------------
#if __has_builtin(__builtin_amdgcn_global_load_async_to_lds_b32) && \
    __has_builtin(__builtin_amdgcn_global_load_async_to_lds_b128)
#define USE_ASYNC_LDS 1
#else
#define USE_ASYNC_LDS 0
#endif

#if USE_ASYNC_LDS
typedef int v4i __attribute__((__vector_size__(16)));
typedef __attribute__((address_space(1))) int* gptr_i32;
typedef __attribute__((address_space(3))) int* lptr_i32;
typedef __attribute__((address_space(1))) v4i* gptr_v4i;
typedef __attribute__((address_space(3))) v4i* lptr_v4i;

static __device__ __forceinline__ void async_cp_b32(const void* g, void* l) {
  __builtin_amdgcn_global_load_async_to_lds_b32((gptr_i32)g, (lptr_i32)l, 0, 0);
}
static __device__ __forceinline__ void async_cp_b128(const void* g, void* l) {
  __builtin_amdgcn_global_load_async_to_lds_b128((gptr_v4i)g, (lptr_v4i)l, 0, 0);
}
#endif

static __device__ __forceinline__ void wait_async0() {
#if __has_builtin(__builtin_amdgcn_s_wait_asynccnt)
  __builtin_amdgcn_s_wait_asynccnt(0);
#else
  asm volatile("s_wait_asynccnt 0" ::: "memory");
#endif
}

// ------------------------------ small kernels ------------------------------

__global__ void cvt_f32_bf16(const float* __restrict__ in,
                             uint16_t* __restrict__ out, int n) {
  int i = blockIdx.x * blockDim.x + threadIdx.x;
  if (i < n) out[i] = f2bf_bits(in[i]);
}

// prod[b,hw] = sum_c x_next[b,c,hw] * x_ref[b,c,hw]
__global__ void prod_kernel(const float* __restrict__ xn,
                            const float* __restrict__ xr,
                            float* __restrict__ prod) {
  int t = blockIdx.x * blockDim.x + threadIdx.x;   // 16384
  int b = t >> 12, hw = t & 4095;
  const float* pn = xn + ((size_t)b << 20) + hw;
  const float* pr = xr + ((size_t)b << 20) + hw;
  float acc = 0.f;
  for (int c = 0; c < 256; ++c)
    acc += pn[(size_t)c << 12] * pr[(size_t)c << 12];
  prod[t] = acc;
}

// corr = leaky_relu(box3x3(prod) / 256)
__global__ void box_lrelu_kernel(const float* __restrict__ prod,
                                 float* __restrict__ corr) {
  int t = blockIdx.x * blockDim.x + threadIdx.x;   // 16384
  int b = t >> 12, hw = t & 4095, h = hw >> 6, w = hw & 63;
  float s = 0.f;
  for (int dy = -1; dy <= 1; ++dy)
    for (int dx = -1; dx <= 1; ++dx) {
      int hh = h + dy, ww = w + dx;
      if (hh >= 0 && hh < 64 && ww >= 0 && ww < 64)
        s += prod[(b << 12) + (hh << 6) + ww];
    }
  s *= (1.f / 256.f);
  corr[t] = s > 0.f ? s : 0.1f * s;
}

// h1 = relu(conv3x3(corr; 1->256) + b1), stored bf16 NCHW
__global__ void conv1_kernel(const float* __restrict__ corr,
                             const float* __restrict__ w1,
                             const float* __restrict__ b1,
                             uint16_t* __restrict__ h1) {
  int t = blockIdx.x * blockDim.x + threadIdx.x;   // 4*256*4096
  int hw = t & 4095, o = (t >> 12) & 255, b = t >> 20;
  int h = hw >> 6, w = hw & 63;
  float acc = b1[o];
  #pragma unroll
  for (int r = 0; r < 9; ++r) {
    int hh = h + r / 3 - 1, ww = w + r % 3 - 1;
    if (hh >= 0 && hh < 64 && ww >= 0 && ww < 64)
      acc += w1[o * 9 + r] * corr[(b << 12) + (hh << 6) + ww];
  }
  h1[t] = f2bf_bits(acc > 0.f ? acc : 0.f);
}

// B matrix producer, K-pair packed: Bp[kp, n] = {bf16(k=2kp), bf16(k=2kp+1)}
// (matches the ISA 16-bit WMMA operand dword layout -> async-copyable slices)
__global__ void im2col_pack_kernel(const uint16_t* __restrict__ h1,
                                   uint32_t* __restrict__ Bp) {
  size_t t = (size_t)blockIdx.x * blockDim.x + threadIdx.x; // 1152*16384
  int n = (int)(t & 16383), kp = (int)(t >> 14);
  int b = n >> 12, hw = n & 4095, h = hw >> 6, w = hw & 63;
  uint32_t packed = 0;
  #pragma unroll
  for (int e = 0; e < 2; ++e) {
    int k = 2 * kp + e;
    int c = k / 9, r = k - c * 9;
    int hh = h + r / 3 - 1, ww = w + r % 3 - 1;
    uint32_t v = 0;
    if (hh >= 0 && hh < 64 && ww >= 0 && ww < 64)
      v = h1[((size_t)(b * 256 + c) << 12) + (hh << 6) + ww];
    packed |= v << (16 * e);
  }
  Bp[t] = packed;
}

// offset = relu(conv3x3(h2; 256->18) + b3)
__global__ void conv3_kernel(const float* __restrict__ h2,
                             const float* __restrict__ w3,
                             const float* __restrict__ b3,
                             float* __restrict__ off) {
  int t = blockIdx.x * blockDim.x + threadIdx.x;   // 4*18*4096
  int hw = t & 4095, j = (t >> 12) % 18, b = (t >> 12) / 18;
  int h = hw >> 6, w = hw & 63;
  float acc = b3[j];
  for (int r = 0; r < 9; ++r) {
    int hh = h + r / 3 - 1, ww = w + r % 3 - 1;
    if (hh < 0 || hh >= 64 || ww < 0 || ww >= 64) continue;
    const float* src = h2 + ((size_t)b << 20) + (hh << 6) + ww;
    const float* wt  = w3 + j * 2304 + r;
    for (int c = 0; c < 256; ++c)
      acc += wt[c * 9] * src[(size_t)c << 12];
  }
  off[t] = acc > 0.f ? acc : 0.f;
}

// Per (b, tap, pixel) bilinear sampling parameters (reused across 256 chans).
__global__ void dcn_params_kernel(const float* __restrict__ off,
                                  int* __restrict__ y0a, int* __restrict__ x0a,
                                  float* __restrict__ wya, float* __restrict__ wxa) {
  int t = blockIdx.x * blockDim.x + threadIdx.x;   // 4*9*4096
  int hw = t & 4095, r = (t >> 12) % 9, b = (t >> 12) / 9;
  int h = hw >> 6, w = hw & 63;
  float dy = off[(((size_t)b * 18 + 2 * r)     << 12) + hw];
  float dx = off[(((size_t)b * 18 + 2 * r + 1) << 12) + hw];
  float y = (float)h + (float)(r / 3 - 1) + dy;
  float x = (float)w + (float)(r % 3 - 1) + dx;
  float y0 = floorf(y), x0 = floorf(x);
  y0a[t] = (int)y0; x0a[t] = (int)x0;
  wya[t] = y - y0;  wxa[t] = x - x0;
}

// DCN B matrix producer, K-pair packed bilinear samples of x_ref.
__global__ void dcn_gather_pack_kernel(const float* __restrict__ xref,
                                       const int* __restrict__ y0a,
                                       const int* __restrict__ x0a,
                                       const float* __restrict__ wya,
                                       const float* __restrict__ wxa,
                                       uint32_t* __restrict__ Bp) {
  size_t t = (size_t)blockIdx.x * blockDim.x + threadIdx.x; // 1152*16384
  int n = (int)(t & 16383), kp = (int)(t >> 14);
  int b = n >> 12, hw = n & 4095;
  uint32_t packed = 0;
  #pragma unroll
  for (int e = 0; e < 2; ++e) {
    int k = 2 * kp + e;
    int c = k / 9, r = k - c * 9;
    int p = ((b * 9 + r) << 12) + hw;
    int y0 = y0a[p], x0 = x0a[p];
    float wy = wya[p], wx = wxa[p];
    const float* src = xref + ((size_t)(b * 256 + c) << 12);
    float v00 = (y0 >= 0 && y0 < 64 && x0     >= 0 && x0     < 64) ? src[( y0      << 6) + x0    ] : 0.f;
    float v01 = (y0 >= 0 && y0 < 64 && x0 + 1 >= 0 && x0 + 1 < 64) ? src[( y0      << 6) + x0 + 1] : 0.f;
    float v10 = (y0 + 1 >= 0 && y0 + 1 < 64 && x0 >= 0 && x0 < 64) ? src[((y0 + 1) << 6) + x0    ] : 0.f;
    float v11 = (y0 + 1 >= 0 && y0 + 1 < 64 && x0 + 1 >= 0 && x0 + 1 < 64) ? src[((y0 + 1) << 6) + x0 + 1] : 0.f;
    float s = v00 * (1 - wy) * (1 - wx) + v01 * (1 - wy) * wx
            + v10 * wy * (1 - wx)       + v11 * wy * wx;
    packed |= ((uint32_t)f2bf_bits(s)) << (16 * e);
  }
  Bp[t] = packed;
}

// ------------------------- WMMA GEMM (shared core) -------------------------
// C[m,n] = sum_k A[m,k]*B[k,n]; M=256, N=16384, K=2304, bf16 in / f32 acc.
// Block: 256 threads (8 waves), tile 128(M) x 128(N); wave = 2x4 16x16 tiles.
// LDS tiles are row-major with pad-20-dword rows (80B, 16B aligned) so each
// lane's WMMA fragment is two contiguous 16B chunks (ds_load_b128 x2).
// mode 0: out = relu(C + bias[m])        (conv2 -> h2, f32 NCHW)
// mode 1: out = 0.5*(xnext + relu(C))    (final output)
#define LPAD 20
__global__ __launch_bounds__(256) void gemm256_wmma(
    const uint16_t* __restrict__ A,    // 256 x 2304 bf16 row-major (weights)
    const uint32_t* __restrict__ Bp,   // 1152 x 16384 K-pair-packed dwords
    const float* __restrict__ bias,
    const float* __restrict__ xnext,
    float* __restrict__ out,
    int mode) {
  __shared__ uint32_t sA[128 * LPAD];  // [m-row][16 K-pair dwords]
  __shared__ uint32_t sB[128 * LPAD];  // [n-row][16 K-pair dwords]

  const int tid  = threadIdx.x;
  const int lane = tid & 31;
  const int wv   = tid >> 5;
  const int l16  = lane & 15;
  const int hi   = lane >> 4;                 // lane half (ISA K-pair layout)
  const int block_m = blockIdx.y * 128;
  const int block_n = blockIdx.x * 128;
  const int mbase = (wv & 3) * 32;
  const int nbase = (wv >> 2) * 64;

  f32x8 acc[2][4];
  #pragma unroll
  for (int mt = 0; mt < 2; ++mt)
    #pragma unroll
    for (int nt = 0; nt < 4; ++nt)
      #pragma unroll
      for (int e = 0; e < 8; ++e) acc[mt][nt][e] = 0.0f;

  for (int k0 = 0; k0 < 2304; k0 += 32) {
    // --- Stage A: 128 rows x 16 dwords, contiguous per row -> b128 copies.
    #pragma unroll
    for (int i = tid; i < 512; i += 256) {
      int row = i >> 2, jc = i & 3;
      const uint32_t* g =
          (const uint32_t*)(A + (size_t)(block_m + row) * 2304 + k0) + jc * 4;
#if USE_ASYNC_LDS
      async_cp_b128(g, &sA[row * LPAD + jc * 4]);
#else
      *(uint4*)&sA[row * LPAD + jc * 4] = *(const uint4*)g;
#endif
    }
    // --- Stage B: 16 pair-rows x 128 n; transpose into n-major LDS rows.
    #pragma unroll
    for (int i = tid; i < 2048; i += 256) {
      int j = i >> 7, n = i & 127;
      const uint32_t* g =
          Bp + (size_t)((k0 >> 1) + j) * 16384 + block_n + n;
#if USE_ASYNC_LDS
      async_cp_b32(g, &sB[n * LPAD + j]);
#else
      sB[n * LPAD + j] = *g;
#endif
    }
#if USE_ASYNC_LDS
    wait_async0();
#endif
    __syncthreads();

    // Fragments: lane row = base + l16; dwords [hi*4 .. hi*4+3] and +8.
    Frag a[2];
    #pragma unroll
    for (int mt = 0; mt < 2; ++mt) {
      const uint32_t* p = &sA[(mbase + mt * 16 + l16) * LPAD + hi * 4];
      uint4 c0 = *(const uint4*)p;
      uint4 c1 = *(const uint4*)(p + 8);
      a[mt].u[0] = c0.x; a[mt].u[1] = c0.y; a[mt].u[2] = c0.z; a[mt].u[3] = c0.w;
      a[mt].u[4] = c1.x; a[mt].u[5] = c1.y; a[mt].u[6] = c1.z; a[mt].u[7] = c1.w;
    }
    #pragma unroll
    for (int nt = 0; nt < 4; ++nt) {
      Frag b;
      const uint32_t* p = &sB[(nbase + nt * 16 + l16) * LPAD + hi * 4];
      uint4 c0 = *(const uint4*)p;
      uint4 c1 = *(const uint4*)(p + 8);
      b.u[0] = c0.x; b.u[1] = c0.y; b.u[2] = c0.z; b.u[3] = c0.w;
      b.u[4] = c1.x; b.u[5] = c1.y; b.u[6] = c1.z; b.u[7] = c1.w;
      #pragma unroll
      for (int mt = 0; mt < 2; ++mt) {
        acc[mt][nt] = __builtin_amdgcn_wmma_f32_16x16x32_bf16(
            false, a[mt].v, false, b.v, (short)0, acc[mt][nt], false, false);
      }
    }
    __syncthreads();
  }

  // Epilogue straight from accumulators (C/D layout: m = v + 8*hi, n = l16).
  #pragma unroll
  for (int mt = 0; mt < 2; ++mt) {
    #pragma unroll
    for (int nt = 0; nt < 4; ++nt) {
      int n = block_n + nbase + nt * 16 + l16;
      int bimg = n >> 12, hw = n & 4095;
      #pragma unroll
      for (int v = 0; v < 8; ++v) {
        int m = block_m + mbase + mt * 16 + v + (hi << 3);
        float val = acc[mt][nt][v];
        size_t oidx = ((size_t)(bimg * 256 + m) << 12) + hw;
        if (mode == 0) {
          val += bias[m];
          out[oidx] = val > 0.f ? val : 0.f;
        } else {
          val = val > 0.f ? val : 0.f;
          out[oidx] = 0.5f * (xnext[oidx] + val);
        }
      }
    }
  }
}

// ------------------------------- launcher ----------------------------------

extern "C" void kernel_launch(void* const* d_in, const int* in_sizes, int n_in,
                              void* d_out, int out_size, void* d_ws, size_t ws_size,
                              hipStream_t stream) {
  const float* x_ref  = (const float*)d_in[0];
  const float* x_next = (const float*)d_in[1];
  const float* w1     = (const float*)d_in[2];
  const float* b1     = (const float*)d_in[3];
  const float* w2     = (const float*)d_in[4];
  const float* b2     = (const float*)d_in[5];
  const float* w3     = (const float*)d_in[6];
  const float* b3     = (const float*)d_in[7];
  const float* w_dcn  = (const float*)d_in[8];
  (void)d_in[9]; (void)in_sizes; (void)n_in; (void)out_size; (void)ws_size;

  char* ws = (char*)d_ws;
  float*    prod  = (float*)   (ws + 0);
  float*    corr  = (float*)   (ws + 65536);
  uint16_t* h1    = (uint16_t*)(ws + 131072);      // 8.39 MB bf16
  uint16_t* w2bf  = (uint16_t*)(ws + 8519680);     // 1.18 MB
  uint16_t* wdbf  = (uint16_t*)(ws + 9699328);     // 1.18 MB
  float*    h2    = (float*)   (ws + 10878976);    // 16.8 MB
  float*    off   = (float*)   (ws + 27656192);    // 1.18 MB
  int*      y0a   = (int*)     (ws + 28835840);
  int*      x0a   = (int*)     (ws + 29425664);
  float*    wya   = (float*)   (ws + 30015488);
  float*    wxa   = (float*)   (ws + 30605312);
  uint32_t* Bp    = (uint32_t*)(ws + 31195136);    // 75.5 MB, reused twice

  // Weight conversion (f32 -> bf16)
  cvt_f32_bf16<<<2304, 256, 0, stream>>>(w2,    w2bf, 589824);
  cvt_f32_bf16<<<2304, 256, 0, stream>>>(w_dcn, wdbf, 589824);

  // correlate
  prod_kernel<<<64, 256, 0, stream>>>(x_next, x_ref, prod);
  box_lrelu_kernel<<<64, 256, 0, stream>>>(prod, corr);

  // conv1 -> h1 (bf16)
  conv1_kernel<<<16384, 256, 0, stream>>>(corr, w1, b1, h1);

  // conv2 as WMMA GEMM: packed im2col then GEMM (+bias, relu) -> h2
  im2col_pack_kernel<<<73728, 256, 0, stream>>>(h1, Bp);
  gemm256_wmma<<<dim3(128, 2), 256, 0, stream>>>(w2bf, Bp, b2, nullptr, h2, 0);

  // conv3 -> offsets
  conv3_kernel<<<1152, 256, 0, stream>>>(h2, w3, b3, off);

  // deformable conv: params, packed bilinear im2col, WMMA GEMM + final mix
  dcn_params_kernel<<<576, 256, 0, stream>>>(off, y0a, x0a, wya, wxa);
  dcn_gather_pack_kernel<<<73728, 256, 0, stream>>>(x_ref, y0a, x0a, wya, wxa, Bp);
  gemm256_wmma<<<dim3(128, 2), 256, 0, stream>>>(wdbf, Bp, nullptr, x_next,
                                                 (float*)d_out, 1);
}
